// MixtralMoE_32066225832168
// MI455X (gfx1250) — matrix operand order repaired and checked by reading, not compiled, not run on hardware
//
#include <hip/hip_runtime.h>
#include <hip/hip_bf16.h>
#include <math.h>

// ---------- CDNA5 WMMA types ----------
typedef __attribute__((ext_vector_type(16))) __bf16 v16bf;
typedef __attribute__((ext_vector_type(8)))  __bf16 v8bf;
typedef __attribute__((ext_vector_type(8)))  float  v8f;

#define E_EXP 8
#define TOPK  2
#define CDIM  1024
#define HDIM  4096
#define NTOK  8192           // B*T = 4*2048
#define CAP   1280           // int(1.25 * 8192 / 8)
#define NSLOT (NTOK * TOPK)

#define KSTEP   32
#define NB      128          // block N tile
#define LDSROW  136          // 128 elements + 8 pad (row stride, 272B, 16B aligned)

__device__ __forceinline__ unsigned short f2bf(float f) {
  union { float f; unsigned u; } c; c.f = f;
  unsigned r = c.u + 0x7FFFu + ((c.u >> 16) & 1u);   // round-to-nearest-even
  return (unsigned short)(r >> 16);
}

// CDNA5 async global->LDS copy of 16 bytes per lane (tracked by ASYNCcnt)
__device__ __forceinline__ void async_copy16(unsigned lds_byte_addr, unsigned long long gaddr) {
  asm volatile("global_load_async_to_lds_b128 %0, %1, off"
               :: "v"(lds_byte_addr), "v"(gaddr) : "memory");
}

// ---------------------------------------------------------------- zero
__global__ void zero_f32(float* __restrict__ p, unsigned long long n) {
  unsigned long long i = (unsigned long long)blockIdx.x * blockDim.x + threadIdx.x;
  unsigned long long s = (unsigned long long)gridDim.x * blockDim.x;
  for (; i < n; i += s) p[i] = 0.0f;
}

// ---------------------------------------------------------------- weight f32 -> bf16
__global__ void convert_weights(const float* __restrict__ w1, const float* __restrict__ w2,
                                unsigned short* __restrict__ w1b, unsigned short* __restrict__ w2b) {
  const unsigned long long n1 = (unsigned long long)E_EXP * CDIM * HDIM;  // 33,554,432
  unsigned long long i = ((unsigned long long)blockIdx.x * blockDim.x + threadIdx.x) * 4ull;
  if (i < n1) {
#pragma unroll
    for (int j = 0; j < 4; ++j) {
      w1b[i + j] = f2bf(w1[i + j]);
      w2b[i + j] = f2bf(w2[i + j]);
    }
  }
}

// ---------------------------------------------------------------- router (1 wave / token)
__global__ void moe_router(const float* __restrict__ xf, const float* __restrict__ rw,
                           const float* __restrict__ rb, int* __restrict__ flat_idx,
                           float* __restrict__ flat_gate, float* __restrict__ imp) {
  const int lane = threadIdx.x & 31;
  const int wave = threadIdx.x >> 5;
  const int n = blockIdx.x * 8 + wave;
  const float* xr = xf + (unsigned long long)n * CDIM;

  float acc[E_EXP];
#pragma unroll
  for (int e = 0; e < E_EXP; ++e) acc[e] = 0.0f;
  for (int c = lane; c < CDIM; c += 32) {
    float xv = xr[c];
#pragma unroll
    for (int e = 0; e < E_EXP; ++e) acc[e] += xv * rw[e * CDIM + c];
  }
#pragma unroll
  for (int e = 0; e < E_EXP; ++e)
#pragma unroll
    for (int off = 16; off; off >>= 1) acc[e] += __shfl_xor(acc[e], off, 32);

  if (lane == 0) {
    float l[E_EXP], p[E_EXP];
    float mx = -1e30f;
#pragma unroll
    for (int e = 0; e < E_EXP; ++e) { l[e] = acc[e] + rb[e]; mx = fmaxf(mx, l[e]); }
    float s = 0.0f;
#pragma unroll
    for (int e = 0; e < E_EXP; ++e) { p[e] = __expf(l[e] - mx); s += p[e]; }
    float inv = 1.0f / s;
#pragma unroll
    for (int e = 0; e < E_EXP; ++e) p[e] *= inv;

    int t1 = 0;
#pragma unroll
    for (int e = 1; e < E_EXP; ++e) if (p[e] > p[t1]) t1 = e;
    int t2 = (t1 == 0) ? 1 : 0;
#pragma unroll
    for (int e = 0; e < E_EXP; ++e) if (e != t1 && p[e] > p[t2]) t2 = e;

    float gs = 1.0f / (p[t1] + p[t2]);
    flat_idx[2 * n]     = t1;
    flat_idx[2 * n + 1] = t2;
    flat_gate[2 * n]     = p[t1] * gs;
    flat_gate[2 * n + 1] = p[t2] * gs;
#pragma unroll
    for (int e = 0; e < E_EXP; ++e) atomicAdd(&imp[e], p[e]);
  }
}

// ---------------------------------------------------------------- FCFS dispatch (1 wave)
__global__ void moe_dispatch(const int* __restrict__ flat_idx, const float* __restrict__ flat_gate,
                             int* __restrict__ tok_buf, float* __restrict__ gate_buf,
                             int* __restrict__ counts) {
  const int lane = threadIdx.x;            // 32 threads, one full wave
  const unsigned below = (1u << lane) - 1u;
  int cnt[E_EXP];
#pragma unroll
  for (int e = 0; e < E_EXP; ++e) cnt[e] = 0;

  for (int s0 = 0; s0 < NSLOT; s0 += 32) {
    const int slot = s0 + lane;
    const int ex = flat_idx[slot];
    const float g = flat_gate[slot];
    int pos = CAP;                          // default: dropped
#pragma unroll
    for (int e = 0; e < E_EXP; ++e) {
      unsigned m32 = (unsigned)__ballot(ex == e);
      if (ex == e) pos = cnt[e] + __popc(m32 & below);
      cnt[e] += __popc(m32);                // uniform across lanes
    }
    if (pos < CAP) {
      int buf = ex * CAP + pos;
      tok_buf[buf]  = slot >> 1;            // owning token
      gate_buf[buf] = g;
    }
  }
  if (lane < E_EXP) counts[lane] = cnt[lane];
}

// ---------------------------------------------------------------- gather x -> bf16 [E,cap,C]
__global__ void moe_gather(const float* __restrict__ xf, const int* __restrict__ tok_buf,
                           unsigned short* __restrict__ inpb) {
  const int slot = blockIdx.x;              // E*CAP blocks
  const int tok = tok_buf[slot];
  const float* src = xf + (unsigned long long)tok * CDIM;
  unsigned short* dst = inpb + (unsigned long long)slot * CDIM;
  const int t = threadIdx.x;
#pragma unroll
  for (int j = 0; j < 4; ++j) {
    int c = t + 256 * j;
    dst[c] = f2bf(src[c]);
  }
}

// ---------------------------------------------------------------- GEMM1: h = gelu(inp @ w1 + b1)
// block = 256 threads (8 waves); block tile 128(M) x 128(N); double-buffered async B staging in LDS
__global__ __launch_bounds__(256)
void moe_gemm1(const unsigned short* __restrict__ inpb, const unsigned short* __restrict__ w1b,
               const float* __restrict__ b1, unsigned short* __restrict__ hb) {
  __shared__ unsigned short smemB[2 * KSTEP * LDSROW];   // 2 x 8704 B

  const int e    = blockIdx.z;
  const int lane = threadIdx.x & 31;
  const int wave = threadIdx.x >> 5;
  const int m0 = blockIdx.y * 128 + wave * 16;
  const int n0 = blockIdx.x * NB;
  const int half = lane >> 4;

  const __bf16* A = reinterpret_cast<const __bf16*>(inpb) + (unsigned long long)e * CAP * CDIM;
  const __bf16* B = reinterpret_cast<const __bf16*>(w1b)  + (unsigned long long)e * CDIM * HDIM;
  const __bf16* Arow = A + (unsigned long long)(m0 + (lane & 15)) * CDIM;

  // async-copy geometry: thread t copies 32B of row (t>>3) of the 32x128 tile
  const int t = threadIdx.x;
  const int crow = t >> 3;                      // 0..31
  const int ccol = (t & 7) * 16;                // element offset within row
  const unsigned ldsbase = (unsigned)(unsigned long long)(&smemB[0]);  // LDS byte addr = low 32 bits
  const unsigned lofs = (unsigned)((crow * LDSROW + ccol) * 2);

  v8f acc[8];
#pragma unroll
  for (int j = 0; j < 8; ++j) acc[j] = v8f{};

  const int KT = CDIM / KSTEP;
  {
    const __bf16* gsrc = B + (unsigned long long)crow * HDIM + n0 + ccol;
    async_copy16(ldsbase + lofs,      (unsigned long long)gsrc);
    async_copy16(ldsbase + lofs + 16, (unsigned long long)(gsrc + 8));
  }
  for (int kt = 0; kt < KT; ++kt) {
    const int buf = kt & 1;
    if (kt + 1 < KT) {
      const __bf16* gsrc = B + (unsigned long long)((kt + 1) * KSTEP + crow) * HDIM + n0 + ccol;
      unsigned dst = ldsbase + (unsigned)(((kt + 1) & 1) * KSTEP * LDSROW * 2) + lofs;
      async_copy16(dst,      (unsigned long long)gsrc);
      async_copy16(dst + 16, (unsigned long long)(gsrc + 8));
      asm volatile("s_wait_asynccnt 0x2" ::: "memory");   // current buffer's copies done
    } else {
      asm volatile("s_wait_asynccnt 0x0" ::: "memory");
    }
    __syncthreads();                                       // all waves' copies landed

    const int k0 = kt * KSTEP;
    union { v16bf v; v8bf h[2]; } af;
    af.h[0] = *reinterpret_cast<const v8bf*>(Arow + k0 + 8 * half);
    af.h[1] = *reinterpret_cast<const v8bf*>(Arow + k0 + 16 + 8 * half);

    const __bf16* bs = reinterpret_cast<const __bf16*>(smemB) + buf * KSTEP * LDSROW + (lane & 15);
#pragma unroll
    for (int j = 0; j < 8; ++j) {
      v16bf bj;
#pragma unroll
      for (int s = 0; s < 16; ++s) bj[s] = bs[((s < 8) ? (8 * half + s) : (16 + 8 * half + s - 8)) * LDSROW + j * 16];
      acc[j] = __builtin_amdgcn_wmma_f32_16x16x32_bf16(false, af.v, false, bj, (short)0, acc[j], false, false);
    }
    __syncthreads();                                       // reads done before buffer reuse
  }

  unsigned short* H = hb + (unsigned long long)e * CAP * HDIM;
  const int mbase = m0 + 8 * half;
  const int col0 = n0 + (lane & 15);
#pragma unroll
  for (int r = 0; r < 8; ++r) {
    const int m = mbase + r;
#pragma unroll
    for (int j = 0; j < 8; ++j) {
      const int col = col0 + 16 * j;
      float v = acc[j][r] + b1[e * HDIM + col];
      float g = 0.5f * v * (1.0f + erff(v * 0.70710678118654752f));   // exact gelu
      H[(unsigned long long)m * HDIM + col] = f2bf(g);
    }
  }
}

// ---------------------------------------------------------------- GEMM2: y += gate*(h @ w2 + b2)
__global__ __launch_bounds__(256)
void moe_gemm2(const unsigned short* __restrict__ hb, const unsigned short* __restrict__ w2b,
               const float* __restrict__ b2, const int* __restrict__ tok_buf,
               const float* __restrict__ gate_buf, float* __restrict__ y) {
  __shared__ unsigned short smemB[2 * KSTEP * LDSROW];

  const int e    = blockIdx.z;
  const int lane = threadIdx.x & 31;
  const int wave = threadIdx.x >> 5;
  const int m0 = blockIdx.y * 128 + wave * 16;
  const int n0 = blockIdx.x * NB;
  const int half = lane >> 4;

  const __bf16* A = reinterpret_cast<const __bf16*>(hb)  + (unsigned long long)e * CAP * HDIM;
  const __bf16* B = reinterpret_cast<const __bf16*>(w2b) + (unsigned long long)e * HDIM * CDIM;
  const __bf16* Arow = A + (unsigned long long)(m0 + (lane & 15)) * HDIM;

  const int t = threadIdx.x;
  const int crow = t >> 3;
  const int ccol = (t & 7) * 16;
  const unsigned ldsbase = (unsigned)(unsigned long long)(&smemB[0]);
  const unsigned lofs = (unsigned)((crow * LDSROW + ccol) * 2);

  v8f acc[8];
#pragma unroll
  for (int j = 0; j < 8; ++j) acc[j] = v8f{};

  const int KT = HDIM / KSTEP;
  {
    const __bf16* gsrc = B + (unsigned long long)crow * CDIM + n0 + ccol;
    async_copy16(ldsbase + lofs,      (unsigned long long)gsrc);
    async_copy16(ldsbase + lofs + 16, (unsigned long long)(gsrc + 8));
  }
  for (int kt = 0; kt < KT; ++kt) {
    const int buf = kt & 1;
    if (kt + 1 < KT) {
      const __bf16* gsrc = B + (unsigned long long)((kt + 1) * KSTEP + crow) * CDIM + n0 + ccol;
      unsigned dst = ldsbase + (unsigned)(((kt + 1) & 1) * KSTEP * LDSROW * 2) + lofs;
      async_copy16(dst,      (unsigned long long)gsrc);
      async_copy16(dst + 16, (unsigned long long)(gsrc + 8));
      asm volatile("s_wait_asynccnt 0x2" ::: "memory");
    } else {
      asm volatile("s_wait_asynccnt 0x0" ::: "memory");
    }
    __syncthreads();

    const int k0 = kt * KSTEP;
    union { v16bf v; v8bf h[2]; } af;
    af.h[0] = *reinterpret_cast<const v8bf*>(Arow + k0 + 8 * half);
    af.h[1] = *reinterpret_cast<const v8bf*>(Arow + k0 + 16 + 8 * half);

    const __bf16* bs = reinterpret_cast<const __bf16*>(smemB) + buf * KSTEP * LDSROW + (lane & 15);
#pragma unroll
    for (int j = 0; j < 8; ++j) {
      v16bf bj;
#pragma unroll
      for (int s = 0; s < 16; ++s) bj[s] = bs[((s < 8) ? (8 * half + s) : (16 + 8 * half + s - 8)) * LDSROW + j * 16];
      acc[j] = __builtin_amdgcn_wmma_f32_16x16x32_bf16(false, af.v, false, bj, (short)0, acc[j], false, false);
    }
    __syncthreads();
  }

  const int mbase = m0 + 8 * half;
  const int col0 = n0 + (lane & 15);
#pragma unroll
  for (int r = 0; r < 8; ++r) {
    const int m = mbase + r;
    const int slot = e * CAP + m;
    const int tok  = tok_buf[slot];
    const float g  = gate_buf[slot];
#pragma unroll
    for (int j = 0; j < 8; ++j) {
      const int col = col0 + 16 * j;
      float v = (acc[j][r] + b2[e * CDIM + col]) * g;
      atomicAdd(&y[(unsigned long long)tok * CDIM + col], v);  // <=2 contributions per token
    }
  }
}

// ---------------------------------------------------------------- aux loss
__global__ void moe_aux(const float* __restrict__ imp, const int* __restrict__ counts,
                        float* __restrict__ out_aux) {
  if (threadIdx.x == 0 && blockIdx.x == 0) {
    float si = 0.0f, sl = 0.0f, ld[E_EXP];
#pragma unroll
    for (int e = 0; e < E_EXP; ++e) si += imp[e];
#pragma unroll
    for (int e = 0; e < E_EXP; ++e) { ld[e] = fminf((float)counts[e], (float)CAP); sl += ld[e]; }
    float aux = 0.0f;
#pragma unroll
    for (int e = 0; e < E_EXP; ++e) aux += (imp[e] / si) * (ld[e] / sl);
    *out_aux = aux * (float)(E_EXP * E_EXP);
  }
}

// ================================================================ host
extern "C" void kernel_launch(void* const* d_in, const int* in_sizes, int n_in,
                              void* d_out, int out_size, void* d_ws, size_t ws_size,
                              hipStream_t stream) {
  const float* x  = (const float*)d_in[0];
  const float* rw = (const float*)d_in[1];
  const float* rb = (const float*)d_in[2];
  const float* w1 = (const float*)d_in[3];
  const float* b1 = (const float*)d_in[4];
  const float* w2 = (const float*)d_in[5];
  const float* b2 = (const float*)d_in[6];
  float* y = (float*)d_out;                           // [NTOK*CDIM] + aux at end

  // workspace layout (bytes)
  char* ws = (char*)d_ws;
  unsigned long long off = 0;
  unsigned short* w1b = (unsigned short*)(ws + off); off += (unsigned long long)E_EXP * CDIM * HDIM * 2; // 64 MB
  unsigned short* w2b = (unsigned short*)(ws + off); off += (unsigned long long)E_EXP * HDIM * CDIM * 2; // 64 MB
  unsigned short* inpb = (unsigned short*)(ws + off); off += (unsigned long long)E_EXP * CAP * CDIM * 2; // 20 MB
  unsigned short* hb   = (unsigned short*)(ws + off); off += (unsigned long long)E_EXP * CAP * HDIM * 2; // 80 MB
  int*   flat_idx  = (int*)(ws + off);   off += (unsigned long long)NSLOT * 4;
  float* flat_gate = (float*)(ws + off); off += (unsigned long long)NSLOT * 4;
  int*   tok_buf   = (int*)(ws + off);   off += (unsigned long long)E_EXP * CAP * 4;   // zero-region start
  float* gate_buf  = (float*)(ws + off); off += (unsigned long long)E_EXP * CAP * 4;
  float* imp       = (float*)(ws + off); off += E_EXP * 4;                             // zero-region end
  int*   counts    = (int*)(ws + off);   off += E_EXP * 4;

  // 1) zero output accumulator and dispatch buffers (poisoned by harness)
  zero_f32<<<2048, 256, 0, stream>>>(y, (unsigned long long)NTOK * CDIM);
  zero_f32<<<80, 256, 0, stream>>>((float*)tok_buf,
                                   (unsigned long long)(E_EXP * CAP * 2 + E_EXP));

  // 2) weights f32 -> bf16
  convert_weights<<<32768, 256, 0, stream>>>(w1, w2, w1b, w2b);

  // 3) router: logits/softmax/top2/gates + importance sums
  moe_router<<<NTOK / 8, 256, 0, stream>>>(x, rw, rb, flat_idx, flat_gate, imp);

  // 4) FCFS capacity dispatch (single wave, exact slot ordering)
  moe_dispatch<<<1, 32, 0, stream>>>(flat_idx, flat_gate, tok_buf, gate_buf, counts);

  // 5) gather token rows into [E,cap,C] bf16
  moe_gather<<<E_EXP * CAP, 256, 0, stream>>>(x, tok_buf, inpb);

  // 6) expert GEMM1 + GeLU  (M=1280, N=4096, K=1024 per expert)
  moe_gemm1<<<dim3(HDIM / NB, CAP / 128, E_EXP), 256, 0, stream>>>(inpb, w1b, b1, hb);

  // 7) expert GEMM2 + gate + scatter-add (M=1280, N=1024, K=4096 per expert)
  moe_gemm2<<<dim3(CDIM / NB, CAP / 128, E_EXP), 256, 0, stream>>>(hb, w2b, b2, tok_buf, gate_buf, y);

  // 8) aux load-balancing loss
  moe_aux<<<1, 32, 0, stream>>>(imp, counts, y + (unsigned long long)NTOK * CDIM);
}
